// NT_Xent_33758442946937
// MI455X (gfx1250) — compile-verified, matrix operand order checked by reading, not run on hardware
//
#include <hip/hip_runtime.h>
#include <hip/hip_bf16.h>

typedef __attribute__((ext_vector_type(16))) _Float16 v16h;
typedef __attribute__((ext_vector_type(8)))  _Float16 v8h;
typedef __attribute__((ext_vector_type(4)))  _Float16 v4h;
typedef __attribute__((ext_vector_type(8)))  float    v8f;

#define NROWS   8192      // N = BATCH * NUM_REPRE
#define DIM     128
#define HALF_N  4096      // BATCH
#define INV_T   10.0f     // 1 / TEMP
#define CSPLIT  4         // column-loop split factor
#define SEGLEN  (NROWS / CSPLIT)   // 2048 columns per segment (power of 2)

// ---------------------------------------------------------------------------
// Kernel 1: row-normalize z = [z1; z2], emit f32 copy (for exact pos term)
// and f16 copy (WMMA operands). One wave per row, 8 waves per block.
// ---------------------------------------------------------------------------
__global__ __launch_bounds__(256) void k_normalize(const float* __restrict__ z1,
                                                   const float* __restrict__ z2,
                                                   float* __restrict__ znf,
                                                   _Float16* __restrict__ znh) {
    const int lane = threadIdx.x & 31;
    const int wave = threadIdx.x >> 5;
    const int row  = blockIdx.x * 8 + wave;          // 1024 blocks -> 8192 rows
    const float* src = (row < HALF_N) ? (z1 + (size_t)row * DIM)
                                      : (z2 + (size_t)(row - HALF_N) * DIM);
    float4 v = ((const float4*)src)[lane];           // 32 lanes x 4 = 128
    float ss = v.x * v.x + v.y * v.y + v.z * v.z + v.w * v.w;
    #pragma unroll
    for (int off = 16; off; off >>= 1) ss += __shfl_xor(ss, off, 32);
    const float inv = 1.0f / fmaxf(sqrtf(ss), 1e-8f);
    v.x *= inv; v.y *= inv; v.z *= inv; v.w *= inv;
    ((float4*)(znf + (size_t)row * DIM))[lane] = v;
    v4h h;
    h.x = (_Float16)v.x; h.y = (_Float16)v.y; h.z = (_Float16)v.z; h.w = (_Float16)v.w;
    ((v4h*)(znh + (size_t)row * DIM))[lane] = h;
}

// ---------------------------------------------------------------------------
// Kernel 2: positive-pair similarity in full f32: pos[i] = dot(zn[i], zn[i^4096]) / T
// ---------------------------------------------------------------------------
__global__ __launch_bounds__(256) void k_pos(const float* __restrict__ znf,
                                             float* __restrict__ posdot) {
    const int lane = threadIdx.x & 31;
    const int wave = threadIdx.x >> 5;
    const int row  = blockIdx.x * 8 + wave;
    const int prt  = row ^ HALF_N;                   // partner column
    float4 a = ((const float4*)(znf + (size_t)row * DIM))[lane];
    float4 b = ((const float4*)(znf + (size_t)prt * DIM))[lane];
    float d = a.x * b.x + a.y * b.y + a.z * b.z + a.w * b.w;
    #pragma unroll
    for (int off = 16; off; off >>= 1) d += __shfl_xor(d, off, 32);
    if (lane == 0) posdot[row] = d * INV_T;
}

// ---------------------------------------------------------------------------
// Kernel 3: fused Gram-matrix + streaming exp-sum (fixed max = 1/T = 10).
// Block (bx, by): rows [64*bx, 64*bx+64), cols [2048*by, 2048*(by+1)).
// 8 waves: 4 row-tiles x 2 column parities; col loop steps by 32, double-
// buffered and unrolled x2 so B loads for tile i+1 overlap WMMAs of tile i.
// Diagonal contribution is NOT masked here; subtracted analytically in k_reduce.
// ---------------------------------------------------------------------------
__global__ __launch_bounds__(256) void k_main(const _Float16* __restrict__ znh,
                                              float* __restrict__ rowpart) {
    __shared__ float lds_sum[64][2];

    const int lane    = threadIdx.x & 31;
    const int wave    = threadIdx.x >> 5;
    const int rowTile = wave >> 1;                   // 0..3
    const int colPar  = wave & 1;                    // 0..1
    const int half    = lane >> 4;                   // K-chunk select (0/1)
    const int ln      = lane & 15;                   // M (for A) / N (for B)
    const int rowBase = blockIdx.x * 64 + rowTile * 16;
    const int colSeg  = blockIdx.y * SEGLEN;

    // --- A fragments, all of K = 128 (ISA 16-bit A 16x32 layout) ---
    v16h A[4];
    {
        const _Float16* arow = znh + (size_t)(rowBase + ln) * DIM;
        #pragma unroll
        for (int c = 0; c < 4; ++c) {
            union { v16h v; v8h h[2]; } u;
            u.h[0] = *(const v8h*)(arow + 32 * c +      8 * half);  // VGPR0-3: K pairs
            u.h[1] = *(const v8h*)(arow + 32 * c + 16 + 8 * half);  // VGPR4-7
            A[c] = u.v;
        }
    }

    float acc[8];
    #pragma unroll
    for (int v = 0; v < 8; ++v) acc[v] = 0.0f;

    union BU { v16h v; v8h h[2]; };
    // Per-lane base: row (colSeg + off + ln), K chunk selected by lane half.
    const _Float16* segBase = znh + (size_t)(colSeg + ln) * DIM + 8 * half;

    // Load one 32-col B panel fragment set at column offset `off` (in segment).
    auto bload = [&](BU* b, int off) {
        const _Float16* p = segBase + (size_t)off * DIM;
        #pragma unroll
        for (int c = 0; c < 4; ++c) {
            b[c].h[0] = *(const v8h*)(p + 32 * c);
            b[c].h[1] = *(const v8h*)(p + 32 * c + 16);
        }
    };
    // 4x WMMA + streaming exp accumulate for one 32-col tile.
    auto compute = [&](const BU* b) {
        v8f C = {};
        #pragma unroll
        for (int c = 0; c < 4; ++c)
            C = __builtin_amdgcn_wmma_f32_16x16x32_f16(
                    false, A[c], false, b[c].v, (short)0, C, false, false);
        #pragma unroll
        for (int v = 0; v < 8; ++v)
            acc[v] += __expf(INV_T * C[v] - INV_T);   // s<=10 => no overflow
    };

    BU b0[4], b1[4];
    int off = 16 * colPar;                  // start offset within segment
    bload(b0, off);
    #pragma unroll 1
    for (int it = 0; it < SEGLEN / 64; ++it) {        // 32 iterations
        bload(b1, (off + 32) & (SEGLEN - 1));         // prefetch next tile
        compute(b0);
        bload(b0, (off + 64) & (SEGLEN - 1));         // prefetch tile after
        compute(b1);
        off += 64;
    }

    // Butterfly-sum the 16 lanes holding each row (stays within half-wave).
    #pragma unroll
    for (int v = 0; v < 8; ++v) {
        float a = acc[v];
        a += __shfl_xor(a, 1, 32);
        a += __shfl_xor(a, 2, 32);
        a += __shfl_xor(a, 4, 32);
        a += __shfl_xor(a, 8, 32);
        acc[v] = a;
    }
    if (ln == 0) {                                    // lanes 0 and 16 write
        #pragma unroll
        for (int v = 0; v < 8; ++v)
            lds_sum[rowTile * 16 + 8 * half + v][colPar] = acc[v];
    }
    __syncthreads();
    if (threadIdx.x < 64) {
        const int r = threadIdx.x;
        rowpart[(size_t)(blockIdx.x * 64 + r) * CSPLIT + blockIdx.y] =
            lds_sum[r][0] + lds_sum[r][1];
    }
}

// ---------------------------------------------------------------------------
// Kernel 4: per-row: combine CSPLIT partials, subtract the diagonal term
// exp(10*<zn16[r],zn16[r]> - 10) (what the WMMA put on the diagonal), take
// lse = 10 + log(sum), minus pos term; fixed-order block reduction -> 32.
// ---------------------------------------------------------------------------
__global__ __launch_bounds__(256) void k_reduce(const float* __restrict__ rowpart,
                                                const _Float16* __restrict__ znh,
                                                const float* __restrict__ posdot,
                                                float* __restrict__ partial) {
    __shared__ float sh[256];
    const int r = blockIdx.x * 256 + threadIdx.x;
    float s = 0.0f;
    #pragma unroll
    for (int j = 0; j < CSPLIT; ++j) s += rowpart[(size_t)r * CSPLIT + j];
    // Diagonal correction: self-dot of the f16 row, accumulated in f32.
    const _Float16* zr = znh + (size_t)r * DIM;
    float d = 0.0f;
    #pragma unroll
    for (int j = 0; j < DIM; j += 8) {
        v8h h = *(const v8h*)(zr + j);
        #pragma unroll
        for (int t = 0; t < 8; ++t) {
            float x = (float)h[t];
            d = fmaf(x, x, d);
        }
    }
    s -= __expf(INV_T * d - INV_T);
    sh[threadIdx.x] = (INV_T + __logf(s)) - posdot[r];
    __syncthreads();
    #pragma unroll
    for (int off = 128; off; off >>= 1) {
        if (threadIdx.x < off) sh[threadIdx.x] += sh[threadIdx.x + off];
        __syncthreads();
    }
    if (threadIdx.x == 0) partial[blockIdx.x] = sh[0];
}

// Kernel 5: final deterministic scalar: sum 32 partials in one wave, / N.
__global__ void k_final(const float* __restrict__ partial, float* __restrict__ out) {
    float v = partial[threadIdx.x];                  // 32 threads
    #pragma unroll
    for (int off = 16; off; off >>= 1) v += __shfl_xor(v, off, 32);
    if (threadIdx.x == 0) out[0] = v * (1.0f / (float)NROWS);
}

// ---------------------------------------------------------------------------
extern "C" void kernel_launch(void* const* d_in, const int* in_sizes, int n_in,
                              void* d_out, int out_size, void* d_ws, size_t ws_size,
                              hipStream_t stream) {
    const float* z1 = (const float*)d_in[0];
    const float* z2 = (const float*)d_in[1];
    float* out = (float*)d_out;

    char* ws = (char*)d_ws;
    float*    znf     = (float*)ws;                                    // 4 MB
    _Float16* znh     = (_Float16*)(ws + (size_t)4 * 1024 * 1024);     // 2 MB
    float*    posdot  = (float*)(ws + (size_t)6 * 1024 * 1024);        // 32 KB
    float*    rowpart = posdot + NROWS;                                // 128 KB
    float*    partial = rowpart + (size_t)NROWS * CSPLIT;              // 128 B

    k_normalize<<<NROWS / 8, 256, 0, stream>>>(z1, z2, znf, znh);
    k_pos<<<NROWS / 8, 256, 0, stream>>>(znf, posdot);
    k_main<<<dim3(NROWS / 64, CSPLIT), 256, 0, stream>>>(znh, rowpart);
    k_reduce<<<NROWS / 256, 256, 0, stream>>>(rowpart, znh, posdot, partial);
    k_final<<<1, 32, 0, stream>>>(partial, out);
}
